// MultiHeadAttention_42339787604655
// MI455X (gfx1250) — compile-verified, matrix operand order checked by reading, not run on hardware
//
#include <hip/hip_runtime.h>
#include <hip/hip_bf16.h>
#include <stdint.h>

// ---------------------------------------------------------------------------
// MI455X (gfx1250) multi-head attention.
//  Pass 0: fp32 -> bf16 conversion of inputs and weights (scale folded in Wq).
//  Pass 1: WMMA bf16 projections  Q,K  [8,4096,64]  and  Vt [8,64,4096].
//  Pass 2: flash attention, TDM (tensor_load_to_lds) double-buffered K/V
//          staging in LDS, bf16 WMMA with f32 accumulate, online softmax.
//  All 16-bit WMMA fragments are loaded as two 16-byte vectors per fragment:
//  lane(half) covers elements [half*8, half*8+8) and [16+half*8, ...+8).
// ---------------------------------------------------------------------------

typedef __bf16    v16bf __attribute__((ext_vector_type(16)));
typedef float     v8f   __attribute__((ext_vector_type(8)));
typedef uint32_t  u32x4 __attribute__((ext_vector_type(4)));
typedef int       i32x8 __attribute__((ext_vector_type(8)));
typedef int       i32x4 __attribute__((ext_vector_type(4)));

#define NQ      4096
#define NKV     4096
#define DIN     1024
#define NHEADS  8
#define HDIM    64

// LDS element (uint16) offsets inside the flash kernel's shared block
#define P_ELEMS   4096                     // P: 8 waves x 16 x 32
#define K_ELEMS   2048                     // K chunk: 32 x 64
#define V_ELEMS   2048                     // V chunk: 64 x 32
#define K_OFF(b)  (P_ELEMS + (b) * K_ELEMS)
#define V_OFF(b)  (P_ELEMS + 2 * K_ELEMS + (b) * V_ELEMS)

union FragB16 { v16bf v; uint32_t u[8]; uint4 q[2]; };

// Load a 16-bit A/B fragment: two contiguous 16B chunks per lane.
//   base: element (uint16) index of this lane's row start; half = lane>>4.
__device__ __forceinline__ void frag_load(FragB16& f, const uint16_t* p,
                                          size_t base, int half) {
    const uint16_t* s = p + base + half * 8;
    f.q[0] = *(const uint4*)(s);
    f.q[1] = *(const uint4*)(s + 16);
}

__device__ __forceinline__ uint32_t f2bf_pack(float lo, float hi) {
    uint32_t ulo = __float_as_uint(lo);
    uint32_t uhi = __float_as_uint(hi);
    ulo = (ulo + 0x7FFFu + ((ulo >> 16) & 1u)) >> 16;
    uhi = (uhi + 0x7FFFu + ((uhi >> 16) & 1u)) >> 16;
    return (ulo & 0xFFFFu) | (uhi << 16);
}
__device__ __forceinline__ uint16_t f2bf(float f) {
    uint32_t u = __float_as_uint(f);
    u = (u + 0x7FFFu + ((u >> 16) & 1u)) >> 16;
    return (uint16_t)u;
}

// ---------------------------------------------------------------------------
// Tensor Data Mover: 2D tile load (bf16 elements) into LDS.
//   tensor: rows of `stride0` elements; tile: tile_d0 x tile_d1 (row-major in
//   LDS, tile_d0 elements per row).  Descriptor per CDNA5 ISA 8.3/8.4.
//   amdgpu-toolchain (clang-23) 6-arg builtin form.
// ---------------------------------------------------------------------------
__device__ __forceinline__ void tdm_load_2d_bf16(uint32_t lds_byte_off, const void* gptr,
                                                 uint32_t tensor_d0, uint32_t tensor_d1,
                                                 uint32_t stride0,
                                                 uint32_t tile_d0, uint32_t tile_d1)
{
    const uint64_t ga = (uint64_t)gptr;
    u32x4 g0;
    g0[0] = 1u;                                            // count=1, is_restore=0
    g0[1] = lds_byte_off;                                  // lds_addr
    g0[2] = (uint32_t)ga;                                  // global_addr[31:0]
    g0[3] = (uint32_t)((ga >> 32) & 0x01FFFFFFu) | (2u << 30); // addr[56:32] | type=2
    i32x8 g1;
    g1[0] = (int)(1u << 16);                               // data_size = 1 (2 bytes)
    g1[1] = (int)(tensor_d0 << 16);                        // tensor_dim0[15:0] @63:48
    g1[2] = (int)((tensor_d0 >> 16) | (tensor_d1 << 16));  // dim0[31:16] | dim1[15:0]
    g1[3] = (int)((tensor_d1 >> 16) | (tile_d0 << 16));    // dim1[31:16] | tile_dim0
    g1[4] = (int)(tile_d1 & 0xFFFFu);                      // tile_dim1, tile_dim2=0
    g1[5] = (int)stride0;                                  // tensor_dim0_stride[31:0]
    g1[6] = 0;                                             // stride0[47:32] | stride1
    g1[7] = 0;
    const i32x4 z4 = {0, 0, 0, 0};                         // 2D: groups 2/3 unused
    const i32x8 z8 = {0, 0, 0, 0, 0, 0, 0, 0};
    __builtin_amdgcn_tensor_load_to_lds(g0, g1, z4, z4, z8, 0);
}

// ---------------------------------------------------------------------------
// Pass 0: fp32 -> bf16 (pairs), optional scale.
// ---------------------------------------------------------------------------
__global__ __launch_bounds__(256)
void cvt_bf16_kernel(const float* __restrict__ src, uint32_t* __restrict__ dst,
                     float scale, int n2)
{
    const int i = blockIdx.x * 256 + threadIdx.x;
    if (i < n2) {
        const float2 f = ((const float2*)src)[i];
        dst[i] = f2bf_pack(f.x * scale, f.y * scale);
    }
}

// ---------------------------------------------------------------------------
// Pass 1: projections (pure bf16 WMMA).  out[h,n,d] = sum_i in[n,i]*W[h,d,i].
// Each wave: 32 rows x 64 dims.  transposed=1 writes [h,d,n] (for V).
// ---------------------------------------------------------------------------
__global__ __launch_bounds__(256)
void mha_proj_kernel(const uint16_t* __restrict__ inBf,  // [4096,1024] bf16
                     const uint16_t* __restrict__ WBf,   // [8,64,1024] bf16
                     uint16_t* __restrict__ outRM,       // [8,4096,64] (or null)
                     uint16_t* __restrict__ outTR,       // [8,64,4096] (or null)
                     int transposed)
{
    const int h    = blockIdx.y;
    const int lane = threadIdx.x & 31;
    const int wave = threadIdx.x >> 5;
    const int n16  = lane & 15;
    const int half = lane >> 4;
    const int rowBase = blockIdx.x * 256 + wave * 32;

    const v8f vzero = {0.f,0.f,0.f,0.f,0.f,0.f,0.f,0.f};
    v8f acc[2][4];
    for (int r = 0; r < 2; ++r)
        for (int dt = 0; dt < 4; ++dt) acc[r][dt] = vzero;

    for (int kk = 0; kk < DIN; kk += 32) {
        FragB16 afrag[2];
        #pragma unroll
        for (int r = 0; r < 2; ++r)
            frag_load(afrag[r], inBf, (size_t)(rowBase + r * 16 + n16) * DIN + kk, half);
        #pragma unroll
        for (int dt = 0; dt < 4; ++dt) {
            FragB16 bfrag;
            frag_load(bfrag, WBf, ((size_t)h * HDIM + dt * 16 + n16) * DIN + kk, half);
            #pragma unroll
            for (int r = 0; r < 2; ++r)
                acc[r][dt] = __builtin_amdgcn_wmma_f32_16x16x32_bf16(
                    false, afrag[r].v, false, bfrag.v,
                    (short)0, acc[r][dt], false, false);
        }
    }

    for (int r = 0; r < 2; ++r) {
        for (int dt = 0; dt < 4; ++dt) {
            #pragma unroll
            for (int v = 0; v < 8; ++v) {
                const int row = rowBase + r * 16 + v + 8 * half;
                const int d   = dt * 16 + n16;
                const uint16_t val = f2bf(acc[r][dt][v]);
                if (!transposed)
                    outRM[((size_t)h * NQ + row) * HDIM + d] = val;
                else
                    outTR[((size_t)h * HDIM + d) * NKV + row] = val;
            }
        }
    }
}

// ---------------------------------------------------------------------------
// Pass 2: flash attention.  Block = 8 waves x 16 query rows; KV chunks of 32
// staged in LDS by the Tensor Data Mover (double buffered).
// LDS map (uint16 units): P[8][16][32] | K0[32][64] K1[32][64] | V0[64][32] V1[64][32]
// ---------------------------------------------------------------------------
__global__ __launch_bounds__(256)
void mha_flash_kernel(const uint16_t* __restrict__ Qb,   // [8,4096,64] (pre-scaled)
                      const uint16_t* __restrict__ Kb,   // [8,4096,64]
                      const uint16_t* __restrict__ Vtb,  // [8,64,4096]
                      float* __restrict__ out)           // [4096,512]
{
    __shared__ uint16_t smem[P_ELEMS + 2 * K_ELEMS + 2 * V_ELEMS];

    const int h    = blockIdx.y;
    const int lane = threadIdx.x & 31;
    const int wave = threadIdx.x >> 5;
    const int n16  = lane & 15;
    const int half = lane >> 4;
    const int qrow0 = blockIdx.x * 128 + wave * 16;

    // Runtime LDS byte offset of smem (LDS aperture: offset = addr[31:0]).
    const uint32_t smem_byte = (uint32_t)(size_t)(const void*)smem;

    const uint16_t* Kh = Kb  + (size_t)h * NKV * HDIM;   // [4096,64]
    const uint16_t* Vh = Vtb + (size_t)h * HDIM * NKV;   // [64,4096]

    // Prologue: DMA chunk 0 (one wave issues; TENSORcnt is per-wave).
    if (wave == 0) {
        tdm_load_2d_bf16(smem_byte + 2u * K_OFF(0), Kh, HDIM, NKV, HDIM, HDIM, 32);
        tdm_load_2d_bf16(smem_byte + 2u * V_OFF(0), Vh, NKV, HDIM, NKV, 32, HDIM);
    }

    // Q fragments: 16 rows x 64 dims = 2 A-fragments, loaded once from global.
    FragB16 qfrag[2];
    {
        const size_t base = ((size_t)h * NQ + (qrow0 + n16)) * HDIM;
        #pragma unroll
        for (int ks = 0; ks < 2; ++ks)
            frag_load(qfrag[ks], Qb, base + ks * 32, half);
    }

    const v8f vzero = {0.f,0.f,0.f,0.f,0.f,0.f,0.f,0.f};
    v8f oacc[4];
    for (int dt = 0; dt < 4; ++dt) oacc[dt] = vzero;
    float rmax[8], rsum[8];
    #pragma unroll
    for (int v = 0; v < 8; ++v) { rmax[v] = -3.0e38f; rsum[v] = 0.0f; }

    uint16_t* myP = smem + wave * (16 * 32);

    for (int kv = 0, it = 0; kv < NKV; kv += 32, ++it) {
        const int cur = it & 1;

        if (wave == 0) __builtin_amdgcn_s_wait_tensorcnt(0);  // chunk `cur` landed
        __syncthreads();

        // Prefetch next chunk into the other buffer while we compute.
        if (wave == 0 && (kv + 32) < NKV) {
            tdm_load_2d_bf16(smem_byte + 2u * K_OFF(cur ^ 1),
                             Kh + (size_t)(kv + 32) * HDIM,
                             HDIM, NKV, HDIM, HDIM, 32);
            tdm_load_2d_bf16(smem_byte + 2u * V_OFF(cur ^ 1),
                             Vh + (kv + 32),
                             NKV, HDIM, NKV, 32, HDIM);
        }

        const uint16_t* Kc = smem + K_OFF(cur);   // [32][64]
        const uint16_t* Vc = smem + V_OFF(cur);   // [64][32]

        // ---- S = Q K^T for 16x32 block (f32 accum) ----
        v8f s[2]; s[0] = vzero; s[1] = vzero;
        #pragma unroll
        for (int t = 0; t < 2; ++t) {
            const int kvrow = t * 16 + n16;            // kv col within chunk
            #pragma unroll
            for (int ks = 0; ks < 2; ++ks) {
                FragB16 bfrag;
                frag_load(bfrag, Kc, (size_t)kvrow * HDIM + ks * 32, half);
                s[t] = __builtin_amdgcn_wmma_f32_16x16x32_bf16(
                    false, qfrag[ks].v, false, bfrag.v, (short)0, s[t], false, false);
            }
        }

        // ---- online softmax (row m = v + 8*half, 32 cols across lane groups) ----
        #pragma unroll
        for (int v = 0; v < 8; ++v) {
            const float a = s[0][v], b = s[1][v];
            float cmax = fmaxf(a, b);
            cmax = fmaxf(cmax, __shfl_xor(cmax, 8, 32));
            cmax = fmaxf(cmax, __shfl_xor(cmax, 4, 32));
            cmax = fmaxf(cmax, __shfl_xor(cmax, 2, 32));
            cmax = fmaxf(cmax, __shfl_xor(cmax, 1, 32));
            const float newm  = fmaxf(rmax[v], cmax);
            const float alpha = __expf(rmax[v] - newm);
            rmax[v] = newm;
            const float p0 = __expf(a - newm);
            const float p1 = __expf(b - newm);
            float psum = p0 + p1;
            psum += __shfl_xor(psum, 8, 32);
            psum += __shfl_xor(psum, 4, 32);
            psum += __shfl_xor(psum, 2, 32);
            psum += __shfl_xor(psum, 1, 32);
            rsum[v] = rsum[v] * alpha + psum;
            #pragma unroll
            for (int dt = 0; dt < 4; ++dt) oacc[dt][v] *= alpha;
            const int mrow = v + 8 * half;
            myP[mrow * 32 + n16]      = f2bf(p0);
            myP[mrow * 32 + 16 + n16] = f2bf(p1);
        }
        asm volatile("s_wait_dscnt 0x0" ::: "memory");   // per-wave LDS RAW

        // ---- reload P as 16x32 A-fragment ----
        FragB16 pfrag;
        frag_load(pfrag, myP, (size_t)n16 * 32, half);

        // ---- O += P * V ----
        #pragma unroll
        for (int dt = 0; dt < 4; ++dt) {
            FragB16 vfrag;
            frag_load(vfrag, Vc, (size_t)(dt * 16 + n16) * 32, half);
            oacc[dt] = __builtin_amdgcn_wmma_f32_16x16x32_bf16(
                false, pfrag.v, false, vfrag.v, (short)0, oacc[dt], false, false);
        }

        __syncthreads();   // all reads of buffer `cur` done before it is reused
    }

    // ---- normalize and write out [4096, 512] f32 ----
    for (int dt = 0; dt < 4; ++dt) {
        #pragma unroll
        for (int v = 0; v < 8; ++v) {
            const int row = qrow0 + v + 8 * half;
            const int col = h * HDIM + dt * 16 + n16;
            out[(size_t)row * (NHEADS * HDIM) + col] = oacc[dt][v] / rsum[v];
        }
    }
}

// ---------------------------------------------------------------------------
extern "C" void kernel_launch(void* const* d_in, const int* in_sizes, int n_in,
                              void* d_out, int out_size, void* d_ws, size_t ws_size,
                              hipStream_t stream) {
    const float* qin = (const float*)d_in[0];
    const float* kin = (const float*)d_in[1];
    const float* vin = (const float*)d_in[2];
    const float* Wqs = (const float*)d_in[3];
    const float* Wks = (const float*)d_in[4];
    const float* Wvs = (const float*)d_in[5];
    float* out = (float*)d_out;

    // Workspace layout (uint16 elements):
    //   Qb, Kb, Vtb        : 3 x 8*4096*64   = 3 x 2M
    //   qinB, kinB, vinB   : 3 x 4096*1024   = 3 x 4M
    //   WqB, WkB, WvB      : 3 x 8*64*1024   = 3 x 512K
    uint16_t* p = (uint16_t*)d_ws;
    const size_t projN = (size_t)NHEADS * NQ * HDIM;
    const size_t inN   = (size_t)NQ * DIN;
    const size_t wN    = (size_t)NHEADS * HDIM * DIN;
    uint16_t* Qb   = p;  p += projN;
    uint16_t* Kb   = p;  p += projN;
    uint16_t* Vtb  = p;  p += projN;
    uint16_t* qinB = p;  p += inN;
    uint16_t* kinB = p;  p += inN;
    uint16_t* vinB = p;  p += inN;
    uint16_t* WqB  = p;  p += wN;
    uint16_t* WkB  = p;  p += wN;
    uint16_t* WvB  = p;  p += wN;

    dim3 blk(256, 1, 1);

    // Pass 0: conversions (1/sqrt(64) folded into Wq).
    const int inN2 = (int)(inN / 2), wN2 = (int)(wN / 2);
    cvt_bf16_kernel<<<dim3((inN2 + 255) / 256), blk, 0, stream>>>(qin, (uint32_t*)qinB, 1.0f,   inN2);
    cvt_bf16_kernel<<<dim3((inN2 + 255) / 256), blk, 0, stream>>>(kin, (uint32_t*)kinB, 1.0f,   inN2);
    cvt_bf16_kernel<<<dim3((inN2 + 255) / 256), blk, 0, stream>>>(vin, (uint32_t*)vinB, 1.0f,   inN2);
    cvt_bf16_kernel<<<dim3((wN2  + 255) / 256), blk, 0, stream>>>(Wqs, (uint32_t*)WqB,  0.125f, wN2);
    cvt_bf16_kernel<<<dim3((wN2  + 255) / 256), blk, 0, stream>>>(Wks, (uint32_t*)WkB,  1.0f,   wN2);
    cvt_bf16_kernel<<<dim3((wN2  + 255) / 256), blk, 0, stream>>>(Wvs, (uint32_t*)WvB,  1.0f,   wN2);

    // Pass 1: projections.
    dim3 gproj(NQ / 256, NHEADS, 1);
    mha_proj_kernel<<<gproj, blk, 0, stream>>>(qinB, WqB, Qb, nullptr, 0);
    mha_proj_kernel<<<gproj, blk, 0, stream>>>(kinB, WkB, Kb, nullptr, 0);
    mha_proj_kernel<<<gproj, blk, 0, stream>>>(vinB, WvB, nullptr, Vtb, 1);

    // Pass 2: flash attention.
    dim3 gfa(NQ / 128, NHEADS, 1);
    mha_flash_kernel<<<gfa, blk, 0, stream>>>(Qb, Kb, Vtb, out);
}
